// CuModel_53893249630597
// MI455X (gfx1250) — matrix operand order repaired and checked by reading, compiled, not run on hardware
//
#include <hip/hip_runtime.h>
#include <hip/hip_bf16.h>

// QuartzNet-style conv stack on MI455X (gfx1250, wave32, WMMA).
// 1x1 convs + k=31 post conv as f16 WMMA GEMMs (f32 accum), pipelined:
// K=64 x N=32 stages, double-buffered LDS, 4 wmma/wave/barrier.

typedef _Float16 half_t;
typedef __attribute__((ext_vector_type(16))) half_t v16h;
typedef __attribute__((ext_vector_type(8)))  half_t v8h;
typedef __attribute__((ext_vector_type(8)))  float  v8f;

#define EPSV 1e-5f

// ---------------------------------------------------------------------------
// Pre: conv1d(1->256, k=9, stride=3, pad=4) + BN + swish, write f16.
// ---------------------------------------------------------------------------
__global__ __launch_bounds__(256)
void pre_kernel(const float* __restrict__ x, const float* __restrict__ w,
                const float* __restrict__ g, const float* __restrict__ b,
                const float* __restrict__ m, const float* __restrict__ v,
                half_t* __restrict__ out, int N, int Co, int Wo, int Win) {
  int id = blockIdx.x * 256 + threadIdx.x;
  if (id >= N * Co * Wo) return;
  int wo = id % Wo;
  int co = (id / Wo) % Co;
  int n  = id / (Wo * Co);
  float s = 0.f;
  int base = 3 * wo - 4;
#pragma unroll
  for (int j = 0; j < 9; ++j) {
    int p = base + j;
    if (p >= 0 && p < Win) s += w[co * 9 + j] * x[(size_t)n * Win + p];
  }
  float inv = g[co] * rsqrtf(v[co] + EPSV);
  float val = s * inv + (b[co] - m[co] * inv);
  val = val / (1.f + __expf(-val));
  out[(size_t)id] = (half_t)val;   // id == ((n*Co)+co)*Wo + wo
}

// ---------------------------------------------------------------------------
// Depthwise conv, 'same' padding. One workgroup per (n, c) row.
// ---------------------------------------------------------------------------
__global__ __launch_bounds__(256)
void dw_kernel(const half_t* __restrict__ in16, const float* __restrict__ dww,
               half_t* __restrict__ out16, int C, int W, int K) {
  __shared__ half_t row[1104];
  __shared__ float  wk[80];
  int n = blockIdx.x / C, c = blockIdx.x % C;
  int tid = threadIdx.x;
  int pad = K >> 1;
  int L = W + K - 1;
  const half_t* src = in16 + ((size_t)n * C + c) * W;
  for (int i = tid; i < L; i += 256) {
    int p = i - pad;
    row[i] = (p >= 0 && p < W) ? src[p] : (half_t)0;
  }
  for (int j = tid; j < K; j += 256) wk[j] = dww[c * K + j];
  __syncthreads();
  for (int wpos = tid; wpos < W; wpos += 256) {
    float s = 0.f;
    for (int j = 0; j < K; ++j) s += (float)row[wpos + j] * wk[j];
    out16[((size_t)n * C + c) * W + wpos] = (half_t)s;
  }
}

// ---------------------------------------------------------------------------
// Weight convert f32 -> f16, relayout (Cout,Cin,KT) -> [t][co][ci].
// ---------------------------------------------------------------------------
__global__ __launch_bounds__(256)
void convw_kernel(const float* __restrict__ w, half_t* __restrict__ out,
                  int Cout, int Cin, int KT) {
  int id = blockIdx.x * 256 + threadIdx.x;
  if (id >= Cout * Cin * KT) return;
  int ci = id % Cin;
  int co = (id / Cin) % Cout;
  int t  = id / (Cin * Cout);
  out[id] = (half_t)w[((size_t)co * Cin + ci) * KT + t];
}

// ---------------------------------------------------------------------------
// WMMA GEMM: D[n,co,p] = post(BN(sum_t sum_ci W16[t,co,ci]*X16[n,ci,p+t-KT/2]))
// Workgroup = 8 waves, tile M=128 x N=32, K staged 64 at a time.
// Double-buffered LDS (2 x 64x32 f16), register-prefetched staging,
// one barrier per stage, 4 wmma per wave per stage.
// ---------------------------------------------------------------------------
__global__ __launch_bounds__(256)
void gemm_wmma(const half_t* __restrict__ in16, const half_t* __restrict__ w16,
               const float* __restrict__ bng, const float* __restrict__ bnb,
               const float* __restrict__ bnm, const float* __restrict__ bnv,
               const float* __restrict__ res32, half_t* __restrict__ out16,
               float* __restrict__ out32, int Cin, int Cout, int W, int KT,
               int do_swish) {
  __shared__ v16h Bsv[256];             // 2 buffers x 2048 halves (8 KB)
  half_t* Bs = (half_t*)Bsv;
  const int tid  = threadIdx.x;
  const int wave = tid >> 5;
  const int lane = tid & 31;
  const int l15  = lane & 15;
  const int hi   = lane >> 4;           // A frag: 0 -> K 0..7/16..23, 1 -> +8
  const int n     = blockIdx.z;
  const int pbase = blockIdx.x * 32;
  const int cow   = blockIdx.y * 128 + wave * 16;
  const int pad   = KT >> 1;
  const size_t inbase = (size_t)n * Cin * W;

  // Staging role: thread t handles k-group kq = t>>5 (8 consecutive k of 64),
  // position pp = t&31. The 8 halves are contiguous in fragment order.
  const int kq = tid >> 5;
  const int pp = tid & 31;
  const int klo = (kq & 3) * 8;                  // 0,8,16,24 within 32-block
  const int soff = ((kq >> 2) * 2 + (pp >> 4)) * 512   // (kb32, pb) region
                 + ((pp & 15) + ((klo & 8) << 1)) * 16 // lane L
                 + ((klo & 16) >> 1);                  // half h base

  const int kchunks = Cin >> 6;
  const int S = KT * kchunks;

  v8f acc0 = {}, acc1 = {};

  // --- prefetch stage 0 into registers ---
  int nt = 0, nkb = 0;
  v8h sreg = {};
  {
    int pos = pbase + pp + nt - pad;
    if (pos >= 0 && pos < W) {
      const half_t* sp = in16 + inbase + (size_t)(nkb + kq * 8) * W + pos;
#pragma unroll
      for (int j = 0; j < 8; ++j) sreg[j] = sp[(size_t)j * W];
    }
  }

  for (int s = 0; s < S; ++s) {
    const int ct = nt, ckb = nkb;
    half_t* buf = Bs + (s & 1) * 2048;
    *(v8h*)(buf + soff) = sreg;

    // prefetch next stage (global loads overlap this stage's wmma)
    nkb += 64;
    if (nkb == Cin) { nkb = 0; ++nt; }
    if (s + 1 < S) {
      v8h r = {};
      int pos = pbase + pp + nt - pad;
      if (pos >= 0 && pos < W) {
        const half_t* sp = in16 + inbase + (size_t)(nkb + kq * 8) * W + pos;
#pragma unroll
        for (int j = 0; j < 8; ++j) r[j] = sp[(size_t)j * W];
      }
      sreg = r;
    }

    __syncthreads();

    // compute: 2 K-blocks x 2 position sub-tiles
    const half_t* wbase = w16 + ((size_t)ct * Cout + cow + l15) * Cin
                              + hi * 8 + ckb;
#pragma unroll
    for (int kb32 = 0; kb32 < 2; ++kb32) {
      const half_t* ap = wbase + kb32 * 32;
      v8h a0 = *(const v8h*)(ap);
      v8h a1 = *(const v8h*)(ap + 16);
      v16h a;
#pragma unroll
      for (int i = 0; i < 8; ++i) { a[i] = a0[i]; a[8 + i] = a1[i]; }
      v16h b0 = *(const v16h*)(buf + (kb32 * 2 + 0) * 512 + lane * 16);
      v16h b1 = *(const v16h*)(buf + (kb32 * 2 + 1) * 512 + lane * 16);
      acc0 = __builtin_amdgcn_wmma_f32_16x16x32_f16(
          false, a, false, b0, (short)0, acc0, false, false);
      acc1 = __builtin_amdgcn_wmma_f32_16x16x32_f16(
          false, a, false, b1, (short)0, acc1, false, false);
    }
  }

  // Epilogue: BN + optional residual + optional swish (both position tiles).
  float inv[8], add[8];
#pragma unroll
  for (int r = 0; r < 8; ++r) {
    int co = cow + r + hi * 8;
    inv[r] = bng[co] * rsqrtf(bnv[co] + EPSV);
    add[r] = bnb[co] - bnm[co] * inv[r];
  }
#pragma unroll
  for (int pb = 0; pb < 2; ++pb) {
    const v8f& acc = pb ? acc1 : acc0;
    int p = pbase + pb * 16 + l15;
    if (p < W) {
#pragma unroll
      for (int r = 0; r < 8; ++r) {
        int co = cow + r + hi * 8;
        float val = acc[r] * inv[r] + add[r];
        size_t o = (size_t)n * Cout * W + (size_t)co * W + p;
        if (res32) val += res32[o];
        if (do_swish) val = val / (1.f + __expf(-val));
        if (out16) out16[o] = (half_t)val;
        if (out32) out32[o] = val;
      }
    }
  }
}

// ---------------------------------------------------------------------------
// Decoder: 768->5 (1x1 conv + bias) + log_softmax over 5, output (N, W, 5).
// ---------------------------------------------------------------------------
__global__ __launch_bounds__(256)
void dec_kernel(const half_t* __restrict__ p16, const float* __restrict__ dw,
                const float* __restrict__ db, float* __restrict__ out,
                int N, int C, int W) {
  int id = blockIdx.x * 256 + threadIdx.x;
  if (id >= N * W) return;
  int n = id / W, wpos = id % W;
  const half_t* base = p16 + (size_t)n * C * W + wpos;
  float lg[5];
#pragma unroll
  for (int c = 0; c < 5; ++c) {
    float s = db[c];
    const float* wc = dw + c * C;
    for (int k = 0; k < C; ++k) s += wc[k] * (float)base[(size_t)k * W];
    lg[c] = s;
  }
  float mx = lg[0];
#pragma unroll
  for (int c = 1; c < 5; ++c) mx = fmaxf(mx, lg[c]);
  float se = 0.f;
#pragma unroll
  for (int c = 0; c < 5; ++c) se += __expf(lg[c] - mx);
  float lse = mx + __logf(se);
#pragma unroll
  for (int c = 0; c < 5; ++c)
    out[((size_t)n * W + wpos) * 5 + c] = lg[c] - lse;
}

// ---------------------------------------------------------------------------
// Host side
// ---------------------------------------------------------------------------
static void run_gemm(hipStream_t stream, const float* wf32, half_t* w16,
                     const half_t* in16, const float* const* bn,
                     const float* res32, half_t* out16, float* out32,
                     int Cin, int Cout, int KT, int swish, int N, int W) {
  int total = Cout * Cin * KT;
  convw_kernel<<<(total + 255) / 256, 256, 0, stream>>>(wf32, w16, Cout, Cin, KT);
  dim3 grid((W + 31) / 32, Cout / 128, N);
  gemm_wmma<<<grid, 256, 0, stream>>>(in16, w16, bn[0], bn[1], bn[2], bn[3],
                                      res32, out16, out32, Cin, Cout, W, KT,
                                      swish);
}

extern "C" void kernel_launch(void* const* d_in, const int* in_sizes, int n_in,
                              void* d_out, int out_size, void* d_ws,
                              size_t ws_size, hipStream_t stream) {
  (void)in_sizes; (void)n_in; (void)out_size; (void)ws_size;
  const int N = 32, W = 1000;
  static const int cfg[5][3] = {
      {256, 256, 33}, {256, 256, 39}, {256, 512, 51}, {512, 512, 63}, {512, 512, 75}};

  int idx = 0;
  const float* x     = (const float*)d_in[idx++];
  const float* pre_w = (const float*)d_in[idx++];
  const float* pre_bn[4];
  for (int i = 0; i < 4; ++i) pre_bn[i] = (const float*)d_in[idx++];

  struct Layer { const float *dw, *pw; const float* bn[4]; };
  struct Blk { Layer lay[5]; const float* res_w; const float* res_bn[4]; };
  Blk blks[5];
  for (int b = 0; b < 5; ++b) {
    for (int r = 0; r < 5; ++r) {
      blks[b].lay[r].dw = (const float*)d_in[idx++];
      blks[b].lay[r].pw = (const float*)d_in[idx++];
      for (int i = 0; i < 4; ++i) blks[b].lay[r].bn[i] = (const float*)d_in[idx++];
    }
    blks[b].res_w = (const float*)d_in[idx++];
    for (int i = 0; i < 4; ++i) blks[b].res_bn[i] = (const float*)d_in[idx++];
  }
  const float* post_w = (const float*)d_in[idx++];
  const float* post_bn[4];
  for (int i = 0; i < 4; ++i) post_bn[i] = (const float*)d_in[idx++];
  const float* dec_w = (const float*)d_in[idx++];
  const float* dec_b = (const float*)d_in[idx++];

  // Workspace layout (bytes):
  char* ws = (char*)d_ws;
  half_t* h16 = (half_t*)(ws + 0);           // 32*512*1000 f16 = 32.768 MB
  half_t* y16 = (half_t*)(ws + 32768000);    // 32.768 MB
  half_t* t16 = (half_t*)(ws + 65536000);    // 32.768 MB
  float*  r32 = (float*) (ws + 98304000);    // 32*512*1000 f32 = 65.536 MB
  half_t* w16 = (half_t*)(ws + 163840000);   // 768*512*31 f16 = 24.379 MB
  half_t* p16 = (half_t*)(ws + 188219392);   // 32*768*1000 f16 = 49.152 MB

  // Pre stage -> h16 (32,256,1000)
  {
    int total = N * 256 * W;
    pre_kernel<<<(total + 255) / 256, 256, 0, stream>>>(
        x, pre_w, pre_bn[0], pre_bn[1], pre_bn[2], pre_bn[3], h16, N, 256, W, 3000);
  }

  for (int b = 0; b < 5; ++b) {
    const int cin = cfg[b][0], cout = cfg[b][1], K = cfg[b][2];
    const half_t* cur = h16;
    int ci = cin;
    for (int r = 0; r < 4; ++r) {
      dw_kernel<<<N * ci, 256, 0, stream>>>(cur, blks[b].lay[r].dw, t16, ci, W, K);
      run_gemm(stream, blks[b].lay[r].pw, w16, t16, blks[b].lay[r].bn,
               nullptr, y16, nullptr, ci, cout, 1, 1, N, W);
      cur = y16; ci = cout;
    }
    // Residual branch: BN(res_w @ h) -> r32 (no swish)
    run_gemm(stream, blks[b].res_w, w16, h16, blks[b].res_bn,
             nullptr, nullptr, r32, cin, cout, 1, 0, N, W);
    // Last repeat (no inner swish): h = swish(BN(pw(dw(y))) + r)
    dw_kernel<<<N * ci, 256, 0, stream>>>(cur, blks[b].lay[4].dw, t16, ci, W, K);
    run_gemm(stream, blks[b].lay[4].pw, w16, t16, blks[b].lay[4].bn,
             r32, h16, nullptr, ci, cout, 1, 1, N, W);
  }

  // Post: conv k=31 (512->768) + BN + swish -> p16
  run_gemm(stream, post_w, w16, h16, post_bn, nullptr, p16, nullptr,
           512, 768, 31, 1, N, W);

  // Decoder + log_softmax -> d_out (N, W, 5) f32
  {
    int total = N * W;
    dec_kernel<<<(total + 255) / 256, 256, 0, stream>>>(
        p16, dec_w, dec_b, (float*)d_out, N, 768, W);
  }
}
